// Cross_PCLEMA_87668872446318
// MI455X (gfx1250) — matrix operand order repaired and checked by reading, not compiled, not run on hardware
//
#include <hip/hip_runtime.h>
#include <math.h>

// ---------------- problem constants ----------------
constexpr int BB = 512;    // batch
constexpr int TT = 32;     // time
constexpr int DD = 512;    // feature dim (K of GEMM1)
constexpr int MM = 1024;   // codebook size (N of GEMM1, K of GEMM2)
constexpr int NN = BB * TT;

// ---------------- WMMA types / helpers ----------------
typedef __attribute__((ext_vector_type(16))) __bf16 v16bf;
typedef __attribute__((ext_vector_type(8)))  __bf16 v8bf;
typedef __attribute__((ext_vector_type(8)))  float  v8f;

__device__ __forceinline__ v16bf cat8(v8bf lo, v8bf hi) {
  return __builtin_shufflevector(lo, hi, 0,1,2,3,4,5,6,7,8,9,10,11,12,13,14,15);
}

// A-fragment 16x32 bf16 (MxK). Lanes 0-15: rows, K 0-7 / 16-23.
// Lanes 16-31: same rows, K 8-15 / 24-31.  (ISA 7.12.2 16-bit A layout)
__device__ __forceinline__ v16bf load_fragA(const __bf16* base, int rowStride, int lane) {
  const int half = lane >> 4, r = lane & 15;
  const __bf16* rp = base + (size_t)r * rowStride + 8 * half;
  v8bf lo = *(const v8bf*)(rp);
  v8bf hi = *(const v8bf*)(rp + 16);
  return cat8(lo, hi);
}

// B-fragment 32x16 (KxN) built from row-major source rows (B = src^T).
// lane&15 = column n (source row n); lanes 0-15 hold K 0-7 / 16-23, lanes 16-31 K 8-15 / 24-31.
__device__ __forceinline__ v16bf load_fragB(const __bf16* base, int rowStride, int lane) {
  const int half = lane >> 4, c = lane & 15;
  const __bf16* rp = base + (size_t)c * rowStride + 8 * half;
  v8bf lo = *(const v8bf*)(rp);
  v8bf hi = *(const v8bf*)(rp + 16);
  return cat8(lo, hi);
}

__device__ __forceinline__ v8f wmma_bf16(v16bf a, v16bf b, v8f c) {
  return __builtin_amdgcn_wmma_f32_16x16x32_bf16(false, a, false, b, (short)0, c, false, false);
}

// ---- CDNA5 async global->LDS copy (ASYNCcnt-tracked, bypasses VGPRs) ----
// lds_off: wave-relative LDS byte address (low 32 bits of the flat pointer,
// per the LDS aperture layout). gptr: per-lane global address (GV mode).
__device__ __forceinline__ void async_copy_b128(unsigned lds_off, const void* gptr) {
  asm volatile("global_load_async_to_lds_b128 %0, %1, off"
               :: "v"(lds_off), "v"(gptr) : "memory");
}
__device__ __forceinline__ void async_wait0() {
  asm volatile("s_wait_asynccnt 0x0" ::: "memory");
}
__device__ __forceinline__ unsigned lds_addr32(const void* p) {
  return (unsigned)(unsigned long long)p;   // flat LDS addr low 32b = LDS offset
}

// fp32 -> bf16 bits, round-to-nearest-even
__device__ __forceinline__ unsigned short f2bf(float x) {
  unsigned int u = __float_as_uint(x);
  unsigned int r = u + 0x7FFFu + ((u >> 16) & 1u);
  return (unsigned short)(r >> 16);
}

// order-preserving float<->uint encode for atomicMin
__device__ __forceinline__ unsigned int fenc(float x) {
  unsigned int u = __float_as_uint(x);
  return (u & 0x80000000u) ? ~u : (u | 0x80000000u);
}
__device__ __forceinline__ float fdec(unsigned int k) {
  unsigned int u = (k & 0x80000000u) ? (k & 0x7FFFFFFFu) : ~k;
  return __uint_as_float(u);
}

// ---------------- k_init: reset min keys ----------------
__global__ void k_init(unsigned int* minkey) {
  if (threadIdx.x < 2) minkey[threadIdx.x] = 0xFFFFFFFFu;
}

// ---------------- k_prep: fp32 -> bf16 + row squared norms ----------------
__global__ void __launch_bounds__(128)
k_prep(const float* __restrict__ a, const float* __restrict__ v, const float* __restrict__ e,
       unsigned short* __restrict__ a_bf, unsigned short* __restrict__ v_bf,
       unsigned short* __restrict__ e_bf,
       float* __restrict__ xx_a, float* __restrict__ xx_v, float* __restrict__ ee)
{
  const int which = blockIdx.y;
  const int row = blockIdx.x;
  const float* src; unsigned short* dst; float* nrm; int nrows;
  if (which == 0)      { src = a; dst = a_bf; nrm = xx_a; nrows = NN; }
  else if (which == 1) { src = v; dst = v_bf; nrm = xx_v; nrows = NN; }
  else                 { src = e; dst = e_bf; nrm = ee;   nrows = MM; }
  if (row >= nrows) return;

  const int tid = threadIdx.x;                       // 128 threads, 4 floats each
  float4 f = ((const float4*)(src + (size_t)row * DD))[tid];
  float s = f.x * f.x + f.y * f.y + f.z * f.z + f.w * f.w;
  unsigned int lo = (unsigned int)f2bf(f.x) | ((unsigned int)f2bf(f.y) << 16);
  unsigned int hi = (unsigned int)f2bf(f.z) | ((unsigned int)f2bf(f.w) << 16);
  ((uint2*)(dst + (size_t)row * DD))[tid] = make_uint2(lo, hi);

  #pragma unroll
  for (int off = 16; off > 0; off >>= 1) s += __shfl_xor(s, off, 32);
  __shared__ float wsum[4];
  if ((tid & 31) == 0) wsum[tid >> 5] = s;
  __syncthreads();
  if (tid == 0) nrm[row] = wsum[0] + wsum[1] + wsum[2] + wsum[3];
}

// ---------------- GEMM1 fused: dist -> sqrt -> dual softmax -> adj/logpH ----
// Grid: (NN/32, 2 modalities), 256 threads (8 waves). Dynamic LDS 128 KB:
//   phase 1: A tile 32x520 bf16 (aliased into Dsh region, barrier separated)
//   phase 2: Dsh 32x1024 fp32 (sqrt distances for row softmax)
__global__ void __launch_bounds__(256)
k_dist_softmax(const unsigned short* __restrict__ a_bf, const unsigned short* __restrict__ v_bf,
               const unsigned short* __restrict__ e_bf,
               const float* __restrict__ xx_a, const float* __restrict__ xx_v,
               const float* __restrict__ ee,
               unsigned short* __restrict__ adj_a, unsigned short* __restrict__ adj_v,
               unsigned short* __restrict__ lp_a,  unsigned short* __restrict__ lp_v)
{
  extern __shared__ __align__(16) char smem[];
  __bf16* Ash = (__bf16*)smem;        // 32 x 520 (padded rows, 16B-aligned stride)
  float*  Dsh = (float*)smem;         // 32 x 1024

  const int mod = blockIdx.y;
  const unsigned short* Xus = mod ? v_bf : a_bf;
  const float* xx           = mod ? xx_v : xx_a;
  unsigned short* adjO      = mod ? adj_v : adj_a;
  unsigned short* lpO       = mod ? lp_v  : lp_a;

  const int n0  = blockIdx.x * 32;
  const int tid = threadIdx.x;

  // stage A tile async: 32 rows x 512 bf16, rows padded to 520 halfwords
  {
    const unsigned ash_base = lds_addr32(Ash);
    for (int idx = tid; idx < 32 * 64; idx += 256) {   // 64 x 16B chunks per row
      const int row = idx >> 6, ch = idx & 63;
      async_copy_b128(ash_base + (unsigned)(row * 1040 + ch * 16),
                      Xus + (size_t)(n0 + row) * DD + ch * 8);
    }
    async_wait0();
  }
  __syncthreads();

  const int wv = tid >> 5, lane = tid & 31;
  const int colBase = wv * 128;                      // each wave: 32 rows x 128 cols
  const __bf16* Ebf = (const __bf16*)e_bf;

  v8f acc[2][8];
  #pragma unroll
  for (int rb = 0; rb < 2; ++rb)
    #pragma unroll
    for (int ct = 0; ct < 8; ++ct) acc[rb][ct] = v8f{};

  for (int kk = 0; kk < DD; kk += 32) {
    v16bf a0 = load_fragA(Ash + kk, 520, lane);
    v16bf a1 = load_fragA(Ash + (size_t)16 * 520 + kk, 520, lane);
    #pragma unroll
    for (int ct = 0; ct < 8; ++ct) {
      const __bf16* bp = Ebf + (size_t)(colBase + ct * 16) * DD + kk;
      v16bf bf = load_fragB(bp, DD, lane);           // e rows straight from L2
      acc[0][ct] = wmma_bf16(a0, bf, acc[0][ct]);
      acc[1][ct] = wmma_bf16(a1, bf, acc[1][ct]);
    }
  }
  __syncthreads();   // all A-tile reads done before aliased Dsh writes

  // epilogue: d = ||x||^2 + ||e||^2 - 2 x.e ; sd = sqrt(max(d,0)) -> LDS
  const int half = lane >> 4, r15 = lane & 15;
  #pragma unroll
  for (int rb = 0; rb < 2; ++rb) {
    #pragma unroll
    for (int ct = 0; ct < 8; ++ct) {
      const int col = colBase + ct * 16 + r15;
      const float eec = ee[col];
      #pragma unroll
      for (int i = 0; i < 8; ++i) {
        const int row = rb * 16 + half * 8 + i;
        float d = xx[n0 + row] + eec - 2.0f * acc[rb][ct][i];
        Dsh[(size_t)row * MM + col] = sqrtf(fmaxf(d, 0.0f));
      }
    }
  }
  __syncthreads();

  // dual softmax per row (t=1 and t=0.5 share one exp: e2 = e1^2)
  for (int rr = 0; rr < 4; ++rr) {
    const int row = wv * 4 + rr;
    const int n = n0 + row;
    const int t = n & (TT - 1);
    const int b = n >> 5;
    const float* drow = Dsh + (size_t)row * MM;

    float mn = 3.4e38f;
    #pragma unroll
    for (int j = 0; j < 32; ++j) mn = fminf(mn, drow[j * 32 + lane]);
    #pragma unroll
    for (int off = 16; off > 0; off >>= 1) mn = fminf(mn, __shfl_xor(mn, off, 32));

    float e1[32];
    float s1 = 0.0f, s2 = 0.0f;
    #pragma unroll
    for (int j = 0; j < 32; ++j) {
      float ev = __expf(mn - drow[j * 32 + lane]);
      e1[j] = ev; s1 += ev; s2 += ev * ev;
    }
    #pragma unroll
    for (int off = 16; off > 0; off >>= 1) {
      s1 += __shfl_xor(s1, off, 32);
      s2 += __shfl_xor(s2, off, 32);
    }
    const float inv1 = 1.0f / s1, inv2 = 1.0f / s2;

    unsigned short* adjRow = adjO + ((size_t)t * BB + b) * MM;  // [t][b][m]
    unsigned short* lpRow  = lpO  + ((size_t)t * BB + b) * MM;
    #pragma unroll
    for (int j = 0; j < 32; ++j) {
      const int col = j * 32 + lane;
      const float ph = e1[j] * inv1;
      adjRow[col] = f2bf((e1[j] * e1[j]) * inv2);
      lpRow[col]  = f2bf(__logf(ph + 1e-10f));
    }
  }
}

// ---------------- GEMM2: S[inst][t] = adj (BxM) @ logpH^T, + global min ----
// Grid: (16 tiles of 128x128, T, 2 insts), 256 threads.
// Double-buffered async global->LDS staging: tile k+1 streams in via the
// Tensor/async path while the 8 WMMAs of tile k execute.
constexpr int TILEB = 128 * 40 * 2;   // one padded 128x32 bf16 tile = 10240 B

__global__ void __launch_bounds__(256)
k_scode(const unsigned short* __restrict__ adj_a, const unsigned short* __restrict__ adj_v,
        const unsigned short* __restrict__ lp_a,  const unsigned short* __restrict__ lp_v,
        float* __restrict__ S, unsigned int* __restrict__ minkey)
{
  extern __shared__ __align__(16) char smem[];
  // [buf][P|Q] tiles, each 128 rows x 40 halfwords (padded)
  __bf16* Ptile[2] = { (__bf16*)(smem + 0 * TILEB), (__bf16*)(smem + 2 * TILEB) };
  __bf16* Qtile[2] = { (__bf16*)(smem + 1 * TILEB), (__bf16*)(smem + 3 * TILEB) };
  float*  red = (float*)(smem + 4 * TILEB);          // 8 wave minima

  const int inst = blockIdx.z;
  const int t = blockIdx.y;
  const int i0 = (blockIdx.x >> 2) * 128;
  const int j0 = (blockIdx.x & 3) * 128;
  const unsigned short* P = (inst ? adj_v : adj_a) + (size_t)t * BB * MM;
  const unsigned short* Q = (inst ? lp_a  : lp_v ) + (size_t)t * BB * MM;
  float* Sout = S + ((size_t)inst * TT + t) * BB * BB;

  const int tid = threadIdx.x;
  const int wv = tid >> 5, lane = tid & 31;
  const int rg = wv & 3, cg = wv >> 2;               // wave: 32 rows x 64 cols

  const int ldrow = tid >> 1, ldh = tid & 1;         // 256 threads: 32B each of P,Q
  const unsigned ldst = (unsigned)(ldrow * 80 + ldh * 32);
  const unsigned pb[2] = { lds_addr32(Ptile[0]), lds_addr32(Ptile[1]) };
  const unsigned qb[2] = { lds_addr32(Qtile[0]), lds_addr32(Qtile[1]) };
  const unsigned short* psrc = P + (size_t)(i0 + ldrow) * MM + ldh * 16;
  const unsigned short* qsrc = Q + (size_t)(j0 + ldrow) * MM + ldh * 16;

  // prologue: stream tile 0
  async_copy_b128(pb[0] + ldst,      psrc);
  async_copy_b128(pb[0] + ldst + 16, psrc + 8);
  async_copy_b128(qb[0] + ldst,      qsrc);
  async_copy_b128(qb[0] + ldst + 16, qsrc + 8);

  v8f acc[2][4];
  #pragma unroll
  for (int rb = 0; rb < 2; ++rb)
    #pragma unroll
    for (int ct = 0; ct < 4; ++ct) acc[rb][ct] = v8f{};

  for (int it = 0; it < MM / 32; ++it) {
    async_wait0();          // current buffer resident (async loads are in-order)
    __syncthreads();        // all waves' copies visible; prev buffer reads done
    if (it + 1 < MM / 32) { // stream next tile into the other buffer
      const int nb = (it + 1) & 1;
      const unsigned short* pn = psrc + (it + 1) * 32;
      const unsigned short* qn = qsrc + (it + 1) * 32;
      async_copy_b128(pb[nb] + ldst,      pn);
      async_copy_b128(pb[nb] + ldst + 16, pn + 8);
      async_copy_b128(qb[nb] + ldst,      qn);
      async_copy_b128(qb[nb] + ldst + 16, qn + 8);
    }
    const __bf16* Pc = Ptile[it & 1];
    const __bf16* Qc = Qtile[it & 1];
    v16bf a0 = load_fragA(Pc + (size_t)(rg * 32 + 0)  * 40, 40, lane);
    v16bf a1 = load_fragA(Pc + (size_t)(rg * 32 + 16) * 40, 40, lane);
    #pragma unroll
    for (int ct = 0; ct < 4; ++ct) {
      v16bf bq = load_fragB(Qc + (size_t)(cg * 64 + ct * 16) * 40, 40, lane);
      acc[0][ct] = wmma_bf16(a0, bq, acc[0][ct]);
      acc[1][ct] = wmma_bf16(a1, bq, acc[1][ct]);
    }
  }

  // write S + tile min
  const int half = lane >> 4, r15 = lane & 15;
  float mn = 3.4e38f;
  #pragma unroll
  for (int rb = 0; rb < 2; ++rb) {
    #pragma unroll
    for (int ct = 0; ct < 4; ++ct) {
      const int col = j0 + cg * 64 + ct * 16 + r15;
      #pragma unroll
      for (int i = 0; i < 8; ++i) {
        const int row = i0 + rg * 32 + rb * 16 + half * 8 + i;
        const float vS = acc[rb][ct][i];
        Sout[(size_t)row * BB + col] = vS;
        mn = fminf(mn, vS);
      }
    }
  }
  #pragma unroll
  for (int off = 16; off > 0; off >>= 1) mn = fminf(mn, __shfl_xor(mn, off, 32));
  if (lane == 0) red[wv] = mn;
  __syncthreads();
  if (tid == 0) {
    float m = red[0];
    #pragma unroll
    for (int k = 1; k < 8; ++k) m = fminf(m, red[k]);
    atomicMin(&minkey[inst], fenc(m));   // order-independent -> deterministic
  }
}

// ---------------- per-row loss terms ----------------
__global__ void __launch_bounds__(256)
k_rowloss(const float* __restrict__ S, const unsigned int* __restrict__ minkey,
          float* __restrict__ term)
{
  const int bx = blockIdx.x;             // 0 .. 2*T*B-1
  const int inst = bx >> 14;
  const int t = (bx >> 9) & 31;
  const int i = bx & 511;
  const float* Srow = S + (((size_t)inst * TT + t) * BB + i) * BB;
  const float minv = fdec(minkey[inst]); // shift = max(-S) = -min(S)

  const int tid = threadIdx.x;
  float s = __expf(Srow[tid] - minv) + __expf(Srow[tid + 256] - minv);
  #pragma unroll
  for (int off = 16; off > 0; off >>= 1) s += __shfl_xor(s, off, 32);
  __shared__ float wsum[8];
  if ((tid & 31) == 0) wsum[tid >> 5] = s;
  __syncthreads();
  if (tid == 0) {
    float tot = 0.0f;
    #pragma unroll
    for (int k = 0; k < 8; ++k) tot += wsum[k];
    term[bx] = (Srow[i] - minv) - __logf(tot + 1e-5f);  // log(diag) - log(denom)
  }
}

// ---------------- deterministic final reduction ----------------
__global__ void __launch_bounds__(256)
k_final(const float* __restrict__ term, float* __restrict__ out)
{
  const int tid = threadIdx.x;
  float s = 0.0f;
  for (int k = tid; k < 2 * TT * BB; k += 256) s += term[k];  // fixed order
  __shared__ float wsum[256];
  wsum[tid] = s;
  __syncthreads();
  for (int off = 128; off > 0; off >>= 1) {
    if (tid < off) wsum[tid] += wsum[tid + off];
    __syncthreads();
  }
  // loss = 0.5*(L1+L2) = -(sum of all terms) / (2*T*B)
  if (tid == 0) out[0] = -wsum[0] / (float)(2 * TT * BB);
}

// ---------------- host launcher ----------------
extern "C" void kernel_launch(void* const* d_in, const int* in_sizes, int n_in,
                              void* d_out, int out_size, void* d_ws, size_t ws_size,
                              hipStream_t stream)
{
  (void)in_sizes; (void)n_in; (void)out_size; (void)ws_size;
  const float* a = (const float*)d_in[0];
  const float* v = (const float*)d_in[1];
  const float* e = (const float*)d_in[2];
  float* out = (float*)d_out;

  char* w = (char*)d_ws;
  auto take = [&](size_t bytes) -> char* {
    char* p = w; w += (bytes + 255) & ~(size_t)255; return p;
  };
  unsigned short* a_bf  = (unsigned short*)take((size_t)NN * DD * 2);
  unsigned short* v_bf  = (unsigned short*)take((size_t)NN * DD * 2);
  unsigned short* e_bf  = (unsigned short*)take((size_t)MM * DD * 2);
  float* xx_a           = (float*)take((size_t)NN * 4);
  float* xx_v           = (float*)take((size_t)NN * 4);
  float* ee             = (float*)take((size_t)MM * 4);
  unsigned short* adj_a = (unsigned short*)take((size_t)TT * BB * MM * 2);
  unsigned short* adj_v = (unsigned short*)take((size_t)TT * BB * MM * 2);
  unsigned short* lp_a  = (unsigned short*)take((size_t)TT * BB * MM * 2);
  unsigned short* lp_v  = (unsigned short*)take((size_t)TT * BB * MM * 2);
  float* S              = (float*)take((size_t)2 * TT * BB * BB * 4);
  unsigned int* minkey  = (unsigned int*)take(256);
  float* term           = (float*)take((size_t)2 * TT * BB * 4);

  k_init<<<1, 32, 0, stream>>>(minkey);
  k_prep<<<dim3(NN, 3), 128, 0, stream>>>(a, v, e, a_bf, v_bf, e_bf, xx_a, xx_v, ee);
  k_dist_softmax<<<dim3(NN / 32, 2), 256, 32 * MM * 4, stream>>>(
      a_bf, v_bf, e_bf, xx_a, xx_v, ee, adj_a, adj_v, lp_a, lp_v);
  k_scode<<<dim3(16, TT, 2), 256, 4 * TILEB + 64, stream>>>(
      adj_a, adj_v, lp_a, lp_v, S, minkey);
  k_rowloss<<<2 * TT * BB, 256, 0, stream>>>(S, minkey, term);
  k_final<<<1, 256, 0, stream>>>(term, out);
}
